// graphEmb_67740224193143
// MI455X (gfx1250) — compile-verified, matrix operand order checked by reading, not run on hardware
//
#include <hip/hip_runtime.h>

#define ALPHA 0.2f
#define NEGF  (-9.0e15f)
#define CDIV(a,b) (((a)+(b)-1)/(b))

typedef __attribute__((ext_vector_type(16))) __bf16 v16bf;
typedef __attribute__((ext_vector_type(8)))  __bf16 v8bf;
typedef __attribute__((ext_vector_type(8)))  float  v8f;

// ---------------------------------------------------------------- utilities
__global__ void k_zero(float* p, int n) {
    int i = blockIdx.x * blockDim.x + threadIdx.x;
    if (i < n) p[i] = 0.0f;
}
__global__ void k_fill_int(int* p, int n, int v) {
    int i = blockIdx.x * blockDim.x + threadIdx.x;
    if (i < n) p[i] = v;
}
__global__ void k_cvt(const float* __restrict__ in, __bf16* __restrict__ out, int n) {
    int i = blockIdx.x * blockDim.x + threadIdx.x;
    if (i < n) out[i] = (__bf16)in[i];
}

__device__ __forceinline__ float dev_sigmoid(float x) { return 1.0f / (1.0f + expf(-x)); }
__device__ __forceinline__ float dev_elu(float x)     { return x > 0.0f ? x : expf(x) - 1.0f; }
__device__ __forceinline__ float dev_leaky(float x)   { return x > 0.0f ? x : ALPHA * x; }

// ---------------------------------------------------- B-operand fragment pack
// Pack a K x N (K%32==0, N%16==0) matrix into WMMA B-fragment order:
// out[((nt*numKt + kt)*32 + lane)*16 + idx]; lane holds column nt*16+(lane&15),
// idx 0-7 -> K = kt*32 + kh + idx, idx 8-15 -> K = kt*32 + kh + 16 + (idx-8),
// kh = (lane>=16)*8  (matches ISA 16-bit 32x16 B layout).
// Og==0 : source element (k,n) = in[k*ld + n]
// Og>0  : combined-head weights [H,K,Og]: (k,n) = in[((n/Og)*K + k)*Og + n%Og]
__global__ void k_pack_b(const float* __restrict__ in, __bf16* __restrict__ out,
                         int K, int N, int ld, int Og,
                         long long inBatch, long long outBatch) {
    in  += (size_t)blockIdx.z * inBatch;
    out += (size_t)blockIdx.z * outBatch;
    int gid = blockIdx.x * blockDim.x + threadIdx.x;
    if (gid >= K * N) return;
    int tile = gid >> 9, r = gid & 511;
    int L = r >> 4, idx = r & 15;
    int numKt = K >> 5;
    int nt = tile / numKt, kt = tile % numKt;
    int n  = nt * 16 + (L & 15);
    int kh = (L >> 4) << 3;
    int k  = kt * 32 + kh + (idx < 8 ? idx : idx + 8);
    float v = (Og > 0) ? in[((size_t)(n / Og) * K + k) * Og + (n % Og)]
                       : in[(size_t)k * ld + n];
    out[gid] = (__bf16)v;
}

// --------------------------------------------- WMMA fragment loads (bf16)
__device__ __forceinline__ v16bf load_a_frag(const __bf16* __restrict__ A, int lda,
                                             int mBase, int kBase, int lane) {
    int m  = mBase + (lane & 15);
    int kh = (lane >> 4) << 3;            // 0 for lanes 0-15, 8 for 16-31
    const __bf16* row = A + (size_t)m * lda + kBase + kh;
    v8bf lo = *(const v8bf*)row;          // K = kh .. kh+7     (16B aligned)
    v8bf hi = *(const v8bf*)(row + 16);   // K = kh+16 .. kh+23
    return __builtin_shufflevector(lo, hi, 0, 1, 2, 3, 4, 5, 6, 7,
                                           8, 9, 10, 11, 12, 13, 14, 15);
}
__device__ __forceinline__ v16bf load_b_packed(const __bf16* __restrict__ Bp, int numKt,
                                               int nBase, int kBase, int lane) {
    const v16bf* t = (const v16bf*)Bp;
    return t[((size_t)(nBase >> 4) * numKt + (kBase >> 5)) * 32 + lane];
}

// --------------------------------------------------------- generic WMMA GEMM
// C[M,N] = A[M,K] @ Bpacked[K,N]; one wave per 16x16 C tile; optional ELU.
__global__ void k_gemm(const __bf16* __restrict__ A, const __bf16* __restrict__ B,
                       float* __restrict__ C, int M, int N, int K,
                       int lda, int ldc,
                       long long aBatch, long long bBatch, long long cBatch, int act) {
    A += (size_t)blockIdx.z * aBatch;
    B += (size_t)blockIdx.z * bBatch;
    C += (size_t)blockIdx.z * cBatch;
    int lane  = threadIdx.x;
    int nBase = blockIdx.x * 16;
    int mBase = blockIdx.y * 16;
    int numKt = K >> 5;
    v8f acc = {};
    for (int kb = 0; kb < K; kb += 32) {
        __builtin_prefetch(A + (size_t)(mBase + (lane & 15)) * lda + kb + 64, 0, 1);
        v16bf a = load_a_frag(A, lda, mBase, kb, lane);
        v16bf b = load_b_packed(B, numKt, nBase, kb, lane);
        acc = __builtin_amdgcn_wmma_f32_16x16x32_bf16(false, a, false, b,
                                                      (short)0, acc, false, false);
    }
    int n    = nBase + (lane & 15);
    int mOff = (lane < 16) ? 0 : 8;
#pragma unroll
    for (int j = 0; j < 8; ++j) {
        float v = acc[j];
        if (act == 1) v = dev_elu(v);
        C[(size_t)(mBase + mOff + j) * ldc + n] = v;
    }
}

// -------------------- fused t[e] = sum_i elu( (edge_attr @ Weg)[e,i] ) * veo[i]
// A = edge_attr bf16 [E,64]; B = Weg packed bf16 [64,2048]; out t[E].
__global__ void k_edge_fold(const __bf16* __restrict__ EA, const __bf16* __restrict__ BW,
                            const float* __restrict__ veo, float* __restrict__ t) {
    int lane  = threadIdx.x;
    int eBase = blockIdx.x * 16;
    float partial[8] = {0.f, 0.f, 0.f, 0.f, 0.f, 0.f, 0.f, 0.f};
    for (int nt = 0; nt < 128; ++nt) {
        v8f acc = {};
#pragma unroll
        for (int kb = 0; kb < 64; kb += 32) {
            v16bf a = load_a_frag(EA, 64, eBase, kb, lane);
            v16bf b = load_b_packed(BW, 2, nt * 16, kb, lane);
            acc = __builtin_amdgcn_wmma_f32_16x16x32_bf16(false, a, false, b,
                                                          (short)0, acc, false, false);
        }
        float vw = veo[nt * 16 + (lane & 15)];
#pragma unroll
        for (int j = 0; j < 8; ++j) partial[j] += dev_elu(acc[j]) * vw;
    }
    __shared__ float red[32][8];
#pragma unroll
    for (int j = 0; j < 8; ++j) red[lane][j] = partial[j];
    __syncthreads();
    if (lane < 16) {
        int j  = (lane < 8) ? lane : lane - 8;
        int l0 = (lane < 8) ? 0 : 16;
        float s = 0.f;
        for (int l = 0; l < 16; ++l) s += red[l0 + l][j];
        t[eBase + lane] = s;
    }
}

// ------------------------------------------------- stage-1 node gating layer
__global__ void k_singlenode(const float* __restrict__ feat, const float* __restrict__ Wsn,
                             const float* __restrict__ asn, float* __restrict__ h0,
                             __bf16* __restrict__ h0b) {
    int n = blockIdx.x, t = threadIdx.x;           // 64 threads
    float wh = 0.f;
    for (int k = 0; k < 64; ++k) wh += feat[n * 64 + k] * Wsn[k * 64 + t];
    __shared__ float sm[64];
    sm[t] = wh * asn[t];
    __syncthreads();
    for (int s = 32; s > 0; s >>= 1) { if (t < s) sm[t] += sm[t + s]; __syncthreads(); }
    float coef = dev_sigmoid(dev_leaky(sm[0]));
    float hv = dev_elu(coef * wh);
    h0[n * 64 + t]  = hv;
    h0b[n * 64 + t] = (__bf16)hv;
}

// -------------------------------------------- global attention (single block)
__global__ void k_gattn(const float* __restrict__ x, const float* __restrict__ W,
                        const float* __restrict__ b, float* __restrict__ out,
                        int N, int F) {
    __shared__ float gate[512];
    __shared__ float sumsh;
    int tid = threadIdx.x;                          // 256 threads
    for (int n = tid; n < N; n += 256) {
        float g = 0.f;
        for (int c = 0; c < F; ++c) g += x[(size_t)n * F + c] * W[c];
        gate[n] = dev_sigmoid(g + b[0]);
    }
    __syncthreads();
    if (tid == 0) {
        float m = -1.0e30f;
        for (int n = 0; n < N; ++n) m = fmaxf(m, gate[n]);
        float s = 0.f;
        for (int n = 0; n < N; ++n) { gate[n] = expf(gate[n] - m); s += gate[n]; }
        sumsh = s;
    }
    __syncthreads();
    float inv = 1.0f / sumsh;
    for (int c = tid; c < F; c += 256) {
        float acc = 0.f;
        for (int n = 0; n < N; ++n) acc += gate[n] * x[(size_t)n * F + c];
        out[c] = acc * inv;
    }
}

// ------------------------------------- source/dest logits es/ed for all heads
__global__ void k_head_logits(const float* __restrict__ Wh, const float* __restrict__ as,
                              const float* __restrict__ ad, float* __restrict__ es,
                              float* __restrict__ ed, int H, int N, int O, int ldw) {
    int gid = blockIdx.x * blockDim.x + threadIdx.x;
    if (gid >= H * N) return;
    int h = gid / N, n = gid % N;
    const float* row = Wh + (size_t)n * ldw + h * O;
    float s = 0.f, d = 0.f;
    for (int o = 0; o < O; ++o) { float w = row[o]; s += w * as[h * O + o]; d += w * ad[h * O + o]; }
    es[gid] = s; ed[gid] = d;
}

// ve[r] = sum_o W[r,o] * a[(r/rowsPerGroup)*O + o]
__global__ void k_foldvec(const float* __restrict__ W, const float* __restrict__ a,
                          float* __restrict__ out, int rows, int O, int rowsPerGroup) {
    int r = blockIdx.x * blockDim.x + threadIdx.x;
    if (r >= rows) return;
    int h = r / rowsPerGroup;
    float s = 0.f;
    for (int o = 0; o < O; ++o) s += W[(size_t)r * O + o] * a[h * O + o];
    out[r] = s;
}

// stage-1 edge logits: ee[h,s,d] = dot(n2n[s*N+d,:I], ve[h,:I]) (dense scatter)
__global__ void k_ee_scatter(const long long* __restrict__ ei, const float* __restrict__ n2n,
                             const float* __restrict__ ve, float* __restrict__ ee,
                             int E, int N, int H, int I) {
    int gid = blockIdx.x * blockDim.x + threadIdx.x;
    if (gid >= E * H) return;
    int e = gid / H, h = gid % H;
    long s = (long)ei[e], d = (long)ei[E + e];
    const float* row = n2n + ((size_t)s * N + d) * I;
    float acc = 0.f;
    for (int k = 0; k < I; ++k) acc += row[k] * ve[h * I + k];
    ee[((size_t)h * N + s) * N + d] = acc;
}

// masked-softmax attention row -> bf16 (block = one (h,s) row, 128 threads)
__global__ void k_attn_row(const float* __restrict__ adj, const float* __restrict__ es,
                           const float* __restrict__ ed, const float* __restrict__ ee,
                           __bf16* __restrict__ attn, int N) {
    int h = blockIdx.x / N, s = blockIdx.x % N;
    int tid = threadIdx.x;
    int C = N >> 7;                                  // cols per thread (<=4)
    const float* eeRow  = ee  + ((size_t)h * N + s) * N;
    const float* adjRow = adj + (size_t)s * N;
    float esv = es[h * N + s];
    float vals[4];
    float vmax = -3.0e38f;
    for (int c = 0; c < C; ++c) {
        int d = tid + c * 128;
        float v = dev_leaky(esv + ed[h * N + d] + eeRow[d]);
        v = (adjRow[d] > 0.f) ? v : NEGF;
        vals[c] = v;
        vmax = fmaxf(vmax, v);
    }
    __shared__ float red[128];
    red[tid] = vmax; __syncthreads();
    for (int st = 64; st > 0; st >>= 1) { if (tid < st) red[tid] = fmaxf(red[tid], red[tid + st]); __syncthreads(); }
    float mx = red[0]; __syncthreads();
    float lsum = 0.f;
    for (int c = 0; c < C; ++c) { vals[c] = expf(vals[c] - mx); lsum += vals[c]; }
    red[tid] = lsum; __syncthreads();
    for (int st = 64; st > 0; st >>= 1) { if (tid < st) red[tid] += red[tid + st]; __syncthreads(); }
    float inv = 1.0f / red[0];
    __bf16* arow = attn + ((size_t)h * N + s) * N;
    for (int c = 0; c < C; ++c) arow[tid + c * 128] = (__bf16)(vals[c] * inv);
}

// deterministic edge pooling on pairs (2p, 2p+1); optional bf16 copy
__global__ void k_edgepool(const float* __restrict__ x, const float* __restrict__ Wp,
                           const float* __restrict__ bp, float* __restrict__ out,
                           __bf16* __restrict__ outb, int F) {
    int p = blockIdx.x, tid = threadIdx.x;          // 128 threads
    const float* x0 = x + (size_t)(2 * p) * F;
    const float* x1 = x0 + F;
    float acc = 0.f;
    for (int j = tid; j < 2 * F; j += 128) acc += (j < F ? x0[j] : x1[j - F]) * Wp[j];
    __shared__ float red[128];
    red[tid] = acc; __syncthreads();
    for (int st = 64; st > 0; st >>= 1) { if (tid < st) red[tid] += red[tid + st]; __syncthreads(); }
    float score = dev_sigmoid(red[0] + bp[0]);
    for (int c = tid; c < F; c += 128) {
        float v = (x0[c] + x1[c]) * score;
        out[(size_t)p * F + c] = v;
        if (outb) outb[(size_t)p * F + c] = (__bf16)v;
    }
}

// stage-2 scatter: adj2 count + "last index wins" ee2 (matches XLA .set)
__global__ void k_scatter_p1(const long long* __restrict__ ei, int* __restrict__ win,
                             float* __restrict__ adj2, int E, int N2) {
    int e = blockIdx.x * blockDim.x + threadIdx.x;
    if (e >= E) return;
    int s2 = (int)(ei[e] >> 1), d2 = (int)(ei[E + e] >> 1);
    int slot = s2 * N2 + d2;
    atomicMax(&win[slot], e);
    atomicAdd(&adj2[slot], 1.0f);
}
__global__ void k_scatter_p2(const long long* __restrict__ ei, const int* __restrict__ win,
                             const float* __restrict__ t, float* __restrict__ ee2,
                             int E, int N2) {
    int e = blockIdx.x * blockDim.x + threadIdx.x;
    if (e >= E) return;
    int s2 = (int)(ei[e] >> 1), d2 = (int)(ei[E + e] >> 1);
    int slot = s2 * N2 + d2;
    if (win[slot] == e) ee2[slot] = t[e];
}

// =============================================================== launch
extern "C" void kernel_launch(void* const* d_in, const int* in_sizes, int n_in,
                              void* d_out, int out_size, void* d_ws, size_t ws_size,
                              hipStream_t stream) {
    const int N = 512, E = 4096, HID = 64, NH = 16, O = 128, CAT = 2048;
    const int N2 = 256;

    const float*     feat = (const float*)d_in[0];
    const long long* ei   = (const long long*)d_in[1];
    const float*     ea   = (const float*)d_in[2];
    const float*     adj  = (const float*)d_in[3];
    const float*     n2n  = (const float*)d_in[4];
    const float* Wsn = (const float*)d_in[5];  const float* asn = (const float*)d_in[6];
    const float* Wg  = (const float*)d_in[7];  const float* Weg = (const float*)d_in[8];
    const float* aS  = (const float*)d_in[9];  const float* aD  = (const float*)d_in[10];
    const float* aE  = (const float*)d_in[11];
    const float* Wp1 = (const float*)d_in[12]; const float* bp1 = (const float*)d_in[13];
    const float* Wo  = (const float*)d_in[14]; const float* Weo = (const float*)d_in[15];
    const float* aoS = (const float*)d_in[16]; const float* aoD = (const float*)d_in[17];
    const float* aoE = (const float*)d_in[18];
    const float* Wp2 = (const float*)d_in[19]; const float* bp2 = (const float*)d_in[20];
    const float* g1W = (const float*)d_in[21]; const float* g1b = (const float*)d_in[22];
    const float* g2W = (const float*)d_in[23]; const float* g2b = (const float*)d_in[24];
    const float* g3W = (const float*)d_in[25]; const float* g3b = (const float*)d_in[26];
    float* out = (float*)d_out;

    // ---- workspace carve
    char* p = (char*)d_ws;
    auto alloc = [&](size_t bytes) { char* r = p; p += (bytes + 255) & ~(size_t)255; return r; };
    float*  h0    = (float*) alloc((size_t)N * HID * 4);
    __bf16* h0b   = (__bf16*)alloc((size_t)N * HID * 2);
    __bf16* Wgb   = (__bf16*)alloc((size_t)HID * CAT * 2);   // packed B
    float*  Whf   = (float*) alloc((size_t)N * CAT * 4);
    __bf16* Whb   = (__bf16*)alloc((size_t)N * CAT * 2);     // packed B, per-head batches
    float*  es    = (float*) alloc((size_t)NH * N * 4);
    float*  ed    = (float*) alloc((size_t)NH * N * 4);
    float*  ve    = (float*) alloc((size_t)NH * HID * 4);
    float*  ee    = (float*) alloc((size_t)NH * N * N * 4);
    __bf16* attnb = (__bf16*)alloc((size_t)NH * N * N * 2);  // row-major A
    float*  hout  = (float*) alloc((size_t)N * CAT * 4);
    float*  hp    = (float*) alloc((size_t)N2 * CAT * 4);
    __bf16* hpb   = (__bf16*)alloc((size_t)N2 * CAT * 2);    // row-major A
    __bf16* Wob   = (__bf16*)alloc((size_t)CAT * O * 2);     // packed B
    __bf16* Bwegb = (__bf16*)alloc((size_t)HID * CAT * 2);   // packed B
    __bf16* eab   = (__bf16*)alloc((size_t)E * HID * 2);     // row-major A
    float*  Wh2   = (float*) alloc((size_t)N2 * O * 4);
    __bf16* Wh2b  = (__bf16*)alloc((size_t)N2 * O * 2);      // packed B
    float*  es2   = (float*) alloc((size_t)N2 * 4);
    float*  ed2   = (float*) alloc((size_t)N2 * 4);
    float*  veo   = (float*) alloc((size_t)CAT * 4);
    float*  tvec  = (float*) alloc((size_t)E * 4);
    float*  adj2  = (float*) alloc((size_t)N2 * N2 * 4);
    float*  ee2   = (float*) alloc((size_t)N2 * N2 * 4);
    int*    win   = (int*)   alloc((size_t)N2 * N2 * 4);
    __bf16* at2b  = (__bf16*)alloc((size_t)N2 * N2 * 2);     // row-major A
    float*  h2    = (float*) alloc((size_t)N2 * O * 4);
    float*  h2p   = (float*) alloc((size_t)(N2 / 2) * O * 4);

    // ---- stage 0: single-node gated projection + global attn 1
    k_singlenode<<<N, 64, 0, stream>>>(feat, Wsn, asn, h0, h0b);
    k_gattn<<<1, 256, 0, stream>>>(h0, g1W, g1b, out + 0, N, HID);

    // ---- stage 1: 16-head GAT
    k_pack_b<<<CDIV(HID * CAT, 256), 256, 0, stream>>>(Wg, Wgb, HID, CAT, 0, O, 0, 0);
    k_gemm<<<dim3(CAT / 16, N / 16, 1), 32, 0, stream>>>(h0b, Wgb, Whf, N, CAT, HID,
                                                          HID, CAT, 0, 0, 0, 0);
    // pack per-head Wh[h] (512x128, row stride CAT) as hout's B operand
    k_pack_b<<<dim3(CDIV(N * O, 256), 1, NH), 256, 0, stream>>>(Whf, Whb, N, O, CAT, 0,
                                                                 O, (long long)N * O);
    k_head_logits<<<CDIV(NH * N, 256), 256, 0, stream>>>(Whf, aS, aD, es, ed, NH, N, O, CAT);
    k_foldvec<<<CDIV(NH * HID, 256), 256, 0, stream>>>(Weg, aE, ve, NH * HID, O, HID);
    k_zero<<<CDIV(NH * N * N, 256), 256, 0, stream>>>(ee, NH * N * N);
    k_ee_scatter<<<CDIV(E * NH, 256), 256, 0, stream>>>(ei, n2n, ve, ee, E, N, NH, HID);
    k_attn_row<<<NH * N, 128, 0, stream>>>(adj, es, ed, ee, attnb, N);
    // hout[h] = attn[h] @ Wh[h], ELU epilogue (concat=True)
    k_gemm<<<dim3(O / 16, N / 16, NH), 32, 0, stream>>>(attnb, Whb, hout, N, O, N,
                                                         N, CAT,
                                                         (long long)N * N, (long long)N * O,
                                                         O, 1);
    k_edgepool<<<N2, 128, 0, stream>>>(hout, Wp1, bp1, hp, hpb, CAT);
    k_gattn<<<1, 256, 0, stream>>>(hp, g2W, g2b, out + HID, N2, CAT);

    // ---- stage 2: single-head GAT on pooled graph
    k_pack_b<<<CDIV(CAT * O, 256), 256, 0, stream>>>(Wo, Wob, CAT, O, O, 0, 0, 0);
    k_gemm<<<dim3(O / 16, N2 / 16, 1), 32, 0, stream>>>(hpb, Wob, Wh2, N2, O, CAT,
                                                         CAT, O, 0, 0, 0, 0);
    k_pack_b<<<CDIV(N2 * O, 256), 256, 0, stream>>>(Wh2, Wh2b, N2, O, O, 0, 0, 0);
    k_head_logits<<<CDIV(N2, 256), 256, 0, stream>>>(Wh2, aoS, aoD, es2, ed2, 1, N2, O, O);
    k_foldvec<<<CDIV(CAT, 256), 256, 0, stream>>>(Weo, aoE, veo, CAT, O, CAT);
    // fused elu(edge_attr @ Weg) . veo  -> per-edge scalar (never materialize [E,2048])
    k_pack_b<<<CDIV(HID * CAT, 256), 256, 0, stream>>>(Weg, Bwegb, HID, CAT, 0, O, 0, 0);
    k_cvt<<<CDIV(E * HID, 256), 256, 0, stream>>>(ea, eab, E * HID);
    k_edge_fold<<<E / 16, 32, 0, stream>>>(eab, Bwegb, veo, tvec);
    // scatter adj2 / ee2 (last-edge-wins, deterministic)
    k_zero<<<CDIV(N2 * N2, 256), 256, 0, stream>>>(adj2, N2 * N2);
    k_zero<<<CDIV(N2 * N2, 256), 256, 0, stream>>>(ee2, N2 * N2);
    k_fill_int<<<CDIV(N2 * N2, 256), 256, 0, stream>>>(win, N2 * N2, -1);
    k_scatter_p1<<<CDIV(E, 256), 256, 0, stream>>>(ei, win, adj2, E, N2);
    k_scatter_p2<<<CDIV(E, 256), 256, 0, stream>>>(ei, win, tvec, ee2, E, N2);
    k_attn_row<<<N2, 128, 0, stream>>>(adj2, es2, ed2, ee2, at2b, N2);
    k_gemm<<<dim3(O / 16, N2 / 16, 1), 32, 0, stream>>>(at2b, Wh2b, h2, N2, O, N2,
                                                         N2, O, 0, 0, 0, 0);
    k_edgepool<<<N2 / 2, 128, 0, stream>>>(h2, Wp2, bp2, h2p, (__bf16*)nullptr, O);
    k_gattn<<<1, 256, 0, stream>>>(h2p, g3W, g3b, out + HID + CAT, N2 / 2, O);
    (void)in_sizes; (void)n_in; (void)out_size; (void)ws_size;
}
